// OPTAttention_57148834840662
// MI455X (gfx1250) — matrix operand order repaired and checked by reading, not compiled, not run on hardware
//
#include <hip/hip_runtime.h>
#include <hip/hip_bf16.h>
#include <math.h>

// ---------------------------------------------------------------------------
// OPT attention with Tender int4 chunked quantization, MI455X (gfx1250).
// All heavy math runs on v_wmma_f32_16x16x32_f16 (wave32 WMMA).
// GEMM uses a double-buffered LDS pipeline fed by GLOBAL_LOAD_ASYNC_TO_LDS
// (ASYNCcnt-tracked) when the toolchain exposes the builtins.
// Workspace layout (requires ws_size >= ~85 MB):
//   [0,32MB)   : 4x dequantized f16 weights, row-major [n][k]
//   [32,40MB)  : quantized activation f16 [2048][2048] (reused for both passes)
//   [40,64MB)  : Q,K,V f16 in [B=2][H=32][T=1024][D=64]
//   [64,80MB)  : attention context f32 [2048][2048]
//   [80MB+..)  : ch_bias / cmax / tmax / 4x CB correction buffers
// ---------------------------------------------------------------------------

typedef __attribute__((ext_vector_type(16))) _Float16 v16h;
typedef __attribute__((ext_vector_type(8)))  _Float16 v8h;
typedef __attribute__((ext_vector_type(8)))  float    v8f;
typedef int v4i __attribute__((vector_size(16)));  // matches builtin signature

#define EMBED   2048
#define MROWS   2048      // B*T
#define CHUNK   256
#define NCHUNK  8         // MROWS / CHUNK
#define HEADS   32
#define HDIM    64
#define SEQ     1024

#if defined(__gfx1250__) &&                                         \
    __has_builtin(__builtin_amdgcn_global_load_async_to_lds_b128) && \
    __has_builtin(__builtin_amdgcn_s_wait_asynccnt)
#define USE_ASYNC_LDS 1
#else
#define USE_ASYNC_LDS 0
#endif

#if USE_ASYNC_LDS
// one wave instruction: each lane copies 16B global -> LDS, no VGPR round-trip
__device__ __forceinline__ void async_cp16(const _Float16* g, _Float16* l) {
  v4i* gg = (v4i*)g;  // C-style: drop const + bitcast (still generic AS)
  v4i* ll = (v4i*)l;
  __builtin_amdgcn_global_load_async_to_lds_b128(
      (__attribute__((address_space(1))) v4i*)gg,
      (__attribute__((address_space(3))) v4i*)ll, 0, 0);
}
#endif

__device__ __forceinline__ v16h ld_frag2(const _Float16* p0, const _Float16* p1) {
  v8h a = *(const v8h*)p0;
  v8h b = *(const v8h*)p1;
  v16h r;
#pragma unroll
  for (int i = 0; i < 8; ++i) { r[i] = a[i]; r[i + 8] = b[i]; }
  return r;
}

__device__ __forceinline__ v8f wmma_f16(v16h a, v16h b, v8f c) {
  return __builtin_amdgcn_wmma_f32_16x16x32_f16(false, a, false, b, (short)0, c,
                                                false, false);
}

// --------------------------- weight int4 qdq --------------------------------
__global__ __launch_bounds__(256) void quant_weight_kernel(
    const float* __restrict__ W, _Float16* __restrict__ Wdq) {
  const int n = blockIdx.x;
  const int tid = threadIdx.x;
  __shared__ float red[256];
  float amax = 0.0f;
  for (int k = tid; k < EMBED; k += 256)
    amax = fmaxf(amax, fabsf(W[(size_t)n * EMBED + k]));
  red[tid] = amax;
  __syncthreads();
  for (int s = 128; s > 0; s >>= 1) {
    if (tid < s) red[tid] = fmaxf(red[tid], red[tid + s]);
    __syncthreads();
  }
  const float scale = fmaxf(red[0] * (1.0f / 7.0f), 1e-9f);
  const float inv = 1.0f / scale;
  for (int k = tid; k < EMBED; k += 256) {
    float w = W[(size_t)n * EMBED + k];
    float q = fminf(fmaxf(rintf(w * inv), -8.0f), 7.0f) * scale;
    Wdq[(size_t)n * EMBED + k] = (_Float16)q;
  }
}

// --------------------- chunk stats: bias + channel max ----------------------
__global__ __launch_bounds__(256) void chunk_stats1_kernel(
    const float* __restrict__ X, float* __restrict__ ch_bias,
    float* __restrict__ cmax) {
  const int c = blockIdx.x >> 3;
  const int h = ((blockIdx.x & 7) << 8) + threadIdx.x;
  const float* p = X + (size_t)(c * CHUNK) * EMBED + h;
  float mx = -1e30f, mn = 1e30f;
  for (int r = 0; r < CHUNK; ++r) {
    float x = p[(size_t)r * EMBED];
    mx = fmaxf(mx, x);
    mn = fminf(mn, x);
  }
  ch_bias[c * EMBED + h] = 0.5f * (mx + mn);
  cmax[c * EMBED + h] = 0.5f * (mx - mn);  // == max|x - bias| exactly
}

__global__ __launch_bounds__(256) void chunk_stats2_kernel(
    const float* __restrict__ cmax, float* __restrict__ tmax) {
  const int c = blockIdx.x;
  const int tid = threadIdx.x;
  __shared__ float red[256];
  float m = 0.0f;
  for (int h = tid; h < EMBED; h += 256) m = fmaxf(m, cmax[c * EMBED + h]);
  red[tid] = m;
  __syncthreads();
  for (int s = 128; s > 0; s >>= 1) {
    if (tid < s) red[tid] = fmaxf(red[tid], red[tid + s]);
    __syncthreads();
  }
  if (tid == 0) tmax[c] = red[0];
}

// ------------------- Tender group-scaled activation qdq ---------------------
__global__ __launch_bounds__(256) void quant_act_kernel(
    const float* __restrict__ X, const float* __restrict__ ch_bias,
    const float* __restrict__ cmax, const float* __restrict__ tmax,
    _Float16* __restrict__ Aq) {
  const int idx = blockIdx.x * 256 + threadIdx.x;
  const int m = idx >> 11;
  const int h = idx & (EMBED - 1);
  const int c = m >> 8;
  const float tm = tmax[c];
  const float cm = cmax[c * EMBED + h];
  int g = 0;
#pragma unroll
  for (int i = 0; i < 7; ++i) {
    float thr = tm * (1.0f / (float)(1 << (7 - i)));
    g += (cm > thr) ? 1 : 0;
  }
  const float s = fmaxf(tm / (float)(1 << (7 - g)) * (1.0f / 7.0f), 1e-9f);
  const float xz = X[idx] - ch_bias[c * EMBED + h];
  const float q = fminf(fmaxf(rintf(xz / s), -8.0f), 7.0f) * s;
  Aq[idx] = (_Float16)q;
}

// ----------------- CB[c][n] = sum_k ch_bias[c][k] * W[n][k] -----------------
__global__ __launch_bounds__(256) void cb_kernel(
    const float* __restrict__ ch_bias, const float* __restrict__ W,
    float* __restrict__ CB) {
  const int n = blockIdx.x;
  const int c = blockIdx.y;
  const int tid = threadIdx.x;
  __shared__ float red[256];
  float s = 0.0f;
  for (int k = tid; k < EMBED; k += 256)
    s += ch_bias[c * EMBED + k] * W[(size_t)n * EMBED + k];
  red[tid] = s;
  __syncthreads();
  for (int o = 128; o > 0; o >>= 1) {
    if (tid < o) red[tid] += red[tid + o];
    __syncthreads();
  }
  if (tid == 0) CB[c * EMBED + n] = red[0];
}

// ---------------------- main WMMA GEMM with fused epilogue ------------------
// Y[m][n] = sum_k A[m][k]*W[n][k] + CB[chunk(m)][n] + bias[n]
// mode 0: f32 [m][n] to outF32.   mode 1: f16 (val*scale) into [B][H][T][D].
__global__ __launch_bounds__(256) void gemm_qdq_kernel(
    const _Float16* __restrict__ A, const _Float16* __restrict__ W,
    const float* __restrict__ CB, const float* __restrict__ bias,
    float* __restrict__ outF32, _Float16* __restrict__ outBHTD, int mode,
    float scale) {
  __shared__ __align__(16) _Float16 As[2][128 * 32];
  __shared__ __align__(16) _Float16 Bs[2][128 * 32];
  const int tid = threadIdx.x;
  const int wid = tid >> 5;
  const int lane = tid & 31;
  const int lm = lane & 15;
  const int hs = lane >> 4;
  const int m0 = blockIdx.y * 128;
  const int n0 = blockIdx.x * 128;
  const int wm = (wid & 3) * 32;   // wave row offset inside block tile
  const int wn = (wid >> 2) * 64;  // wave col offset inside block tile

  v8f acc[2][4];
#pragma unroll
  for (int i = 0; i < 2; ++i)
#pragma unroll
    for (int j = 0; j < 4; ++j)
#pragma unroll
      for (int e = 0; e < 8; ++e) acc[i][j][e] = 0.0f;

  const int srow = tid >> 1;          // 0..127 staging row
  const int scol = (tid & 1) * 16;    // 0 or 16
  const _Float16* gA = A + (size_t)(m0 + srow) * EMBED + scol;
  const _Float16* gB = W + (size_t)(n0 + srow) * EMBED + scol;
  const int lofs = srow * 32 + scol;

  int buf = 0;
#if USE_ASYNC_LDS
  // prologue: fill buffer 0 for k0 = 0 (4 async b128 per lane => ASYNCcnt = 4)
  async_cp16(gA,     &As[0][lofs]);
  async_cp16(gA + 8, &As[0][lofs + 8]);
  async_cp16(gB,     &Bs[0][lofs]);
  async_cp16(gB + 8, &Bs[0][lofs + 8]);
#endif

  for (int k0 = 0; k0 < EMBED; k0 += 32) {
#if USE_ASYNC_LDS
    if (k0 + 32 < EMBED) {
      // overlap: fill the other buffer while current tile computes
      const _Float16* nA = gA + k0 + 32;
      const _Float16* nB = gB + k0 + 32;
      async_cp16(nA,     &As[buf ^ 1][lofs]);
      async_cp16(nA + 8, &As[buf ^ 1][lofs + 8]);
      async_cp16(nB,     &Bs[buf ^ 1][lofs]);
      async_cp16(nB + 8, &Bs[buf ^ 1][lofs + 8]);
      __builtin_amdgcn_s_wait_asynccnt(4);  // current tile's 4 are complete
    } else {
      __builtin_amdgcn_s_wait_asynccnt(0);
    }
    __syncthreads();  // other waves' async fills of current buffer visible
#else
    {  // synchronous staging fallback (single logical buffer, 'buf' fixed 0)
      if (k0 + 32 < EMBED) __builtin_prefetch(gA + k0 + 32, 0, 1);
      const _Float16* pa = gA + k0;
      *(v8h*)&As[buf][lofs] = *(const v8h*)pa;
      *(v8h*)&As[buf][lofs + 8] = *(const v8h*)(pa + 8);
      const _Float16* pb = gB + k0;
      *(v8h*)&Bs[buf][lofs] = *(const v8h*)pb;
      *(v8h*)&Bs[buf][lofs + 8] = *(const v8h*)(pb + 8);
    }
    __syncthreads();
#endif

    v16h af[2], bf[4];
#pragma unroll
    for (int i = 0; i < 2; ++i) {
      // A layout: lane row = lm, K segments hs*8..+7 and hs*8+16..+23
      const int row = wm + 16 * i + lm;
      af[i] = ld_frag2(&As[buf][row * 32 + hs * 8],
                       &As[buf][row * 32 + hs * 8 + 16]);
    }
#pragma unroll
    for (int j = 0; j < 4; ++j) {
      // B layout: lane col = lm, K segments hs*8..+7 and hs*8+16..+23
      const int row = wn + 16 * j + lm;
      bf[j] = ld_frag2(&Bs[buf][row * 32 + hs * 8],
                       &Bs[buf][row * 32 + hs * 8 + 16]);
    }
#pragma unroll
    for (int i = 0; i < 2; ++i)
#pragma unroll
      for (int j = 0; j < 4; ++j) acc[i][j] = wmma_f16(af[i], bf[j], acc[i][j]);
    __syncthreads();  // nobody still reads this buffer when it is refilled
#if USE_ASYNC_LDS
    buf ^= 1;
#endif
  }

  // epilogue (C layout: row = v + 8*hs, col = lm)
#pragma unroll
  for (int i = 0; i < 2; ++i)
#pragma unroll
    for (int j = 0; j < 4; ++j) {
      const int mb = m0 + wm + 16 * i;
      const int n = n0 + wn + 16 * j + lm;
      const float bn = bias[n];
#pragma unroll
      for (int v = 0; v < 8; ++v) {
        const int m = mb + v + 8 * hs;
        const float val = acc[i][j][v] + CB[(m >> 8) * EMBED + n] + bn;
        if (mode == 0) {
          outF32[(size_t)m * EMBED + n] = val;
        } else {
          const int b = m >> 10, t = m & (SEQ - 1);
          const int h = n >> 6, d = n & (HDIM - 1);
          outBHTD[((((size_t)(b * HEADS + h)) * SEQ + t) << 6) + d] =
              (_Float16)(val * scale);
        }
      }
    }
}

// ------------------------- flash attention (WMMA) ---------------------------
// grid.x = T/128 q tiles, grid.y = B*HEADS. Q pre-scaled by 1/sqrt(HDIM).
__global__ __launch_bounds__(256) void flash_attn_kernel(
    const _Float16* __restrict__ Q, const _Float16* __restrict__ K,
    const _Float16* __restrict__ V, float* __restrict__ ctx) {
  const int bh = blockIdx.y;
  const int q0 = blockIdx.x * 128;
  const size_t base = (size_t)bh * SEQ * HDIM;
  const int tid = threadIdx.x;
  const int wid = tid >> 5;
  const int lane = tid & 31;
  const int lm = lane & 15;
  const int hs = lane >> 4;

  __shared__ __align__(16) _Float16 Qs[128 * 64];
  __shared__ __align__(16) _Float16 Ks[64 * 64];
  __shared__ __align__(16) _Float16 Vt[64 * 64];   // [d][key]
  __shared__ __align__(16) _Float16 Ps[8 * 16 * 64];  // per-wave P scratch

  {  // stage Q tile 128x64
    const int r = tid >> 1, c0 = (tid & 1) * 32;
    const _Float16* gp = Q + base + (size_t)(q0 + r) * HDIM + c0;
#pragma unroll
    for (int s = 0; s < 4; ++s)
      *(v8h*)&Qs[r * 64 + c0 + s * 8] = *(const v8h*)(gp + s * 8);
  }
  __syncthreads();

  v16h aq[2];
#pragma unroll
  for (int c = 0; c < 2; ++c) {
    const int row = wid * 16 + lm;
    const int kb = c * 32 + hs * 8;
    aq[c] = ld_frag2(&Qs[row * 64 + kb], &Qs[row * 64 + kb + 16]);
  }

  float m_i[8], l_i[8];
  v8f o[4];
#pragma unroll
  for (int v = 0; v < 8; ++v) { m_i[v] = -1e30f; l_i[v] = 0.0f; }
#pragma unroll
  for (int j = 0; j < 4; ++j)
#pragma unroll
    for (int e = 0; e < 8; ++e) o[j][e] = 0.0f;

  const int sr = tid >> 2, sc = (tid & 3) * 16;  // staging coords for K/V
  for (int j0 = 0; j0 < SEQ; j0 += 64) {
    __syncthreads();  // protect Ks/Vt against previous iteration readers
    {  // stage K tile [key][d] (already B-layout friendly)
      const _Float16* gp = K + base + (size_t)(j0 + sr) * HDIM + sc;
      *(v8h*)&Ks[sr * 64 + sc] = *(const v8h*)gp;
      *(v8h*)&Ks[sr * 64 + sc + 8] = *(const v8h*)(gp + 8);
    }
    {  // stage V transposed: Vt[d][key]
      const _Float16* gp = V + base + (size_t)(j0 + sr) * HDIM + sc;
      v8h d0 = *(const v8h*)gp;
      v8h d1 = *(const v8h*)(gp + 8);
#pragma unroll
      for (int e = 0; e < 8; ++e) {
        Vt[(sc + e) * 64 + sr] = d0[e];
        Vt[(sc + 8 + e) * 64 + sr] = d1[e];
      }
    }
    if (j0 + 64 < SEQ) {  // prefetch next tiles into GL2 (global_prefetch_b8)
      __builtin_prefetch(K + base + (size_t)(j0 + 64 + sr) * HDIM + sc, 0, 1);
      __builtin_prefetch(V + base + (size_t)(j0 + 64 + sr) * HDIM + sc, 0, 1);
    }
    __syncthreads();

    // S = Q * K^T for this wave's 16 rows x 64 keys
    v8f s[4];
#pragma unroll
    for (int j = 0; j < 4; ++j)
#pragma unroll
      for (int e = 0; e < 8; ++e) s[j][e] = 0.0f;
#pragma unroll
    for (int j = 0; j < 4; ++j)
#pragma unroll
      for (int c = 0; c < 2; ++c) {
        const int row = 16 * j + lm;
        const int kb = c * 32 + hs * 8;
        v16h bk = ld_frag2(&Ks[row * 64 + kb], &Ks[row * 64 + kb + 16]);
        s[j] = wmma_f16(aq[c], bk, s[j]);
      }

    // online softmax, rows m = v + 8*hs, cols across 16 lanes of each half
#pragma unroll
    for (int v = 0; v < 8; ++v) {
      float mx = fmaxf(fmaxf(s[0][v], s[1][v]), fmaxf(s[2][v], s[3][v]));
#pragma unroll
      for (int off = 1; off < 16; off <<= 1)
        mx = fmaxf(mx, __shfl_xor(mx, off, 32));
      const float mn = fmaxf(m_i[v], mx);
      const float alpha = __expf(m_i[v] - mn);
      m_i[v] = mn;
#pragma unroll
      for (int j = 0; j < 4; ++j) s[j][v] = __expf(s[j][v] - mn);
      float rs = s[0][v] + s[1][v] + s[2][v] + s[3][v];
#pragma unroll
      for (int off = 1; off < 16; off <<= 1) rs += __shfl_xor(rs, off, 32);
      l_i[v] = l_i[v] * alpha + rs;
#pragma unroll
      for (int j = 0; j < 4; ++j) o[j][v] *= alpha;
    }

    // C layout -> A layout through per-wave LDS scratch
#pragma unroll
    for (int j = 0; j < 4; ++j)
#pragma unroll
      for (int v = 0; v < 8; ++v)
        Ps[(wid * 16 + v + 8 * hs) * 64 + 16 * j + lm] = (_Float16)s[j][v];

    // O += P * V
#pragma unroll
    for (int c = 0; c < 2; ++c) {
      const int kb = c * 32 + hs * 8;
      const int prow = wid * 16 + lm;
      v16h ap = ld_frag2(&Ps[prow * 64 + kb], &Ps[prow * 64 + kb + 16]);
#pragma unroll
      for (int j = 0; j < 4; ++j) {
        const int kb2 = c * 32 + hs * 8;
        v16h bv = ld_frag2(&Vt[(16 * j + lm) * 64 + kb2],
                           &Vt[(16 * j + lm) * 64 + kb2 + 16]);
        o[j] = wmma_f16(ap, bv, o[j]);
      }
    }
  }

  // epilogue: ctx[b*SEQ + row][h*64 + d] = o / l
  const int b = bh >> 5, h = bh & 31;
#pragma unroll
  for (int j = 0; j < 4; ++j)
#pragma unroll
    for (int v = 0; v < 8; ++v) {
      const int row = q0 + wid * 16 + v + 8 * hs;
      const int d = 16 * j + lm;
      ctx[(size_t)(b * SEQ + row) * EMBED + h * HDIM + d] = o[j][v] / l_i[v];
    }
}

// ---------------------------------------------------------------------------
extern "C" void kernel_launch(void* const* d_in, const int* in_sizes, int n_in,
                              void* d_out, int out_size, void* d_ws,
                              size_t ws_size, hipStream_t stream) {
  (void)in_sizes; (void)n_in; (void)out_size; (void)ws_size;
  const float* hidden = (const float*)d_in[0];
  const float* wq = (const float*)d_in[1];
  const float* wk = (const float*)d_in[2];
  const float* wv = (const float*)d_in[3];
  const float* wo = (const float*)d_in[4];
  const float* bq = (const float*)d_in[5];
  const float* bk = (const float*)d_in[6];
  const float* bv = (const float*)d_in[7];
  const float* bo = (const float*)d_in[8];

  char* ws = (char*)d_ws;
  const size_t MB = 1024 * 1024;
  _Float16* WdqQ = (_Float16*)(ws + 0 * MB);
  _Float16* WdqK = (_Float16*)(ws + 8 * MB);
  _Float16* WdqV = (_Float16*)(ws + 16 * MB);
  _Float16* WdqO = (_Float16*)(ws + 24 * MB);
  _Float16* Aq   = (_Float16*)(ws + 32 * MB);
  _Float16* Qh   = (_Float16*)(ws + 40 * MB);
  _Float16* Kh   = (_Float16*)(ws + 48 * MB);
  _Float16* Vh   = (_Float16*)(ws + 56 * MB);
  float* ctx     = (float*)(ws + 64 * MB);
  float* chb     = (float*)(ws + 80 * MB);
  float* cmax    = (float*)(ws + 80 * MB + 64 * 1024);
  float* tmax    = (float*)(ws + 80 * MB + 128 * 1024);
  float* CBq     = (float*)(ws + 80 * MB + 192 * 1024);
  float* CBk     = (float*)(ws + 80 * MB + 256 * 1024);
  float* CBv     = (float*)(ws + 80 * MB + 320 * 1024);
  float* CBo     = (float*)(ws + 80 * MB + 384 * 1024);

  // 1) int4 qdq all four weights -> f16 [n][k]
  quant_weight_kernel<<<EMBED, 256, 0, stream>>>(wq, WdqQ);
  quant_weight_kernel<<<EMBED, 256, 0, stream>>>(wk, WdqK);
  quant_weight_kernel<<<EMBED, 256, 0, stream>>>(wv, WdqV);
  quant_weight_kernel<<<EMBED, 256, 0, stream>>>(wo, WdqO);

  // 2) chunk decomposition of hidden_states
  chunk_stats1_kernel<<<NCHUNK * 8, 256, 0, stream>>>(hidden, chb, cmax);
  chunk_stats2_kernel<<<NCHUNK, 256, 0, stream>>>(cmax, tmax);
  quant_act_kernel<<<(MROWS * EMBED) / 256, 256, 0, stream>>>(hidden, chb, cmax,
                                                              tmax, Aq);
  cb_kernel<<<dim3(EMBED, NCHUNK), 256, 0, stream>>>(chb, wq, CBq);
  cb_kernel<<<dim3(EMBED, NCHUNK), 256, 0, stream>>>(chb, wk, CBk);
  cb_kernel<<<dim3(EMBED, NCHUNK), 256, 0, stream>>>(chb, wv, CBv);

  // 3) Q/K/V projections (Q fused with 1/sqrt(64) scaling), f16 [B][H][T][D]
  dim3 ggrid(EMBED / 128, MROWS / 128);
  gemm_qdq_kernel<<<ggrid, 256, 0, stream>>>(Aq, WdqQ, CBq, bq, nullptr, Qh, 1,
                                             0.125f);
  gemm_qdq_kernel<<<ggrid, 256, 0, stream>>>(Aq, WdqK, CBk, bk, nullptr, Kh, 1,
                                             1.0f);
  gemm_qdq_kernel<<<ggrid, 256, 0, stream>>>(Aq, WdqV, CBv, bv, nullptr, Vh, 1,
                                             1.0f);

  // 4) flash attention -> ctx f32 [2048][2048]
  flash_attn_kernel<<<dim3(SEQ / 128, 2 * HEADS), 256, 0, stream>>>(Qh, Kh, Vh,
                                                                    ctx);

  // 5) output projection through the same quant pipeline -> d_out (f32)
  chunk_stats1_kernel<<<NCHUNK * 8, 256, 0, stream>>>(ctx, chb, cmax);
  chunk_stats2_kernel<<<NCHUNK, 256, 0, stream>>>(cmax, tmax);
  quant_act_kernel<<<(MROWS * EMBED) / 256, 256, 0, stream>>>(ctx, chb, cmax,
                                                              tmax, Aq);
  cb_kernel<<<dim3(EMBED, NCHUNK), 256, 0, stream>>>(chb, wo, CBo);
  gemm_qdq_kernel<<<ggrid, 256, 0, stream>>>(Aq, WdqO, CBo, bo, (float*)d_out,
                                             nullptr, 0, 1.0f);
}
